// Qwen2LlambaBlock_62130996904035
// MI455X (gfx1250) — matrix operand order repaired and checked by reading, not compiled, not run on hardware
//
#include <hip/hip_runtime.h>
#include <cstdint>

// ---------------------------------------------------------------------------
// Qwen2 Llamba block for MI455X (gfx1250, wave32).
// GEMMs: bf16 WMMA 16x16x32, f32 accumulate; K-tiles staged into LDS by the
//        Tensor Data Mover (tensor_load_to_lds), double-buffered, with the
//        TDM pad feature generating a bank-conflict-free 80B LDS row stride.
// Scan:  register-resident 64x64 state per (batch, head).
// ---------------------------------------------------------------------------

typedef __bf16 bf16;
typedef __attribute__((ext_vector_type(16))) __bf16 v16bf;
typedef __attribute__((ext_vector_type(8)))  __bf16 v8bf;
typedef __attribute__((ext_vector_type(8)))  float  v8f;
typedef __attribute__((ext_vector_type(4)))  unsigned v4u;
typedef __attribute__((ext_vector_type(8)))  int   v8i;
typedef __attribute__((ext_vector_type(4)))  int   v4i;

#define BATCH    2
#define SEQ      2048
#define MTOK     4096      // BATCH*SEQ
#define DMODEL   2048
#define NHEADS   32
#define DSTATE   64
#define HEADD    64
#define CONVD    6144      // DMODEL + 2*NHEADS*DSTATE
#define DINP     8224      // 2*DMODEL + 2*NHEADS*DSTATE + NHEADS
#define DINP_PAD 8320      // round up to 128
#define INTER_N  5504      // multiple of 128 already
#define LDSS     40        // LDS row stride in bf16 (32 data + 8 pad = 80B)

__device__ __forceinline__ bf16 f2bf(float f) {
  unsigned u = __float_as_uint(f);
  unsigned short r = (unsigned short)((u + 0x7FFFu + ((u >> 16) & 1u)) >> 16);
  return __builtin_bit_cast(bf16, r);
}

__device__ __forceinline__ v16bf combine16(v8bf lo, v8bf hi) {
  v16bf f;
#pragma unroll
  for (int i = 0; i < 8; ++i) { f[i] = lo[i]; f[i + 8] = hi[i]; }
  return f;
}

// Wave-relative LDS byte offset of a __shared__ object (generic -> AS3 -> int).
__device__ __forceinline__ unsigned lds_addr_of(const void* p) {
  return (unsigned)(uintptr_t)(__attribute__((address_space(3))) const char*)p;
}

// ---------------------------------------------------------------------------
// TDM: DMA one 128-row x 32-col bf16 tile (row stride K elements in memory)
// into LDS at lds_byte, emitting 16B of padding after every 64B row so the
// LDS image has an 80B (LDSS) row stride.
//   group1: data_size=1 (2B), pad_enable, pad_interval=3 (16 DW), pad_amount=3
//           (4 DW), tensor_dim0=K, tensor_dim1=rows_total, tile 32x128.
// ---------------------------------------------------------------------------
__device__ __forceinline__ void tdm_load_tile(unsigned lds_byte,
                                              const bf16* gtile,
                                              unsigned K, unsigned rows_total) {
  unsigned long long ga = (unsigned long long)(uintptr_t)gtile;
  v4u g0;
  g0[0] = 1u;                                             // count=1, user mode
  g0[1] = lds_byte;                                       // lds_addr
  g0[2] = (unsigned)(ga & 0xFFFFFFFFu);                   // global_addr lo
  g0[3] = (unsigned)((ga >> 32) & 0x01FFFFFFu) | (2u << 30); // addr hi, type=2
  v8i g1;
  g1[0] = (int)((1u << 16) | (1u << 20) | (3u << 22) | (3u << 25));
  g1[1] = (int)((K & 0xFFFFu) << 16);                     // tensor_dim0 lo16
  g1[2] = (int)((K >> 16) | ((rows_total & 0xFFFFu) << 16)); // dim0 hi | dim1 lo
  g1[3] = (int)((rows_total >> 16) | (32u << 16));        // dim1 hi | tile_dim0=32
  g1[4] = 128;                                            // tile_dim1=128, tile_dim2=0
  g1[5] = (int)K;                                         // tensor_dim0_stride lo32
  g1[6] = 0;                                              // stride hi, dim1_stride lo
  g1[7] = 0;
  v4i zz4 = {0, 0, 0, 0};
  v8i zz8 = {0, 0, 0, 0, 0, 0, 0, 0};
  __builtin_amdgcn_tensor_load_to_lds(g0, g1, zz4, zz4, zz8, 0);
}

// ---------------------------------------------------------------------------
// f32 -> bf16 weight conversion with zero-padding of rows N..Npad-1.
// ---------------------------------------------------------------------------
__global__ void cvt_pad_kernel(const float* __restrict__ src, bf16* __restrict__ dst,
                               int N, int K, int Npad) {
  size_t i = (size_t)blockIdx.x * 256 + threadIdx.x;
  size_t tot = (size_t)Npad * (size_t)K;
  if (i >= tot) return;
  int n = (int)(i / (size_t)K);
  dst[i] = (n < N) ? f2bf(src[i]) : __builtin_bit_cast(bf16, (unsigned short)0);
}

// ---------------------------------------------------------------------------
// RMSNorm over D=2048, optional per-token mask, bf16 output.
// ---------------------------------------------------------------------------
__global__ __launch_bounds__(256) void rmsnorm_kernel(
    const float* __restrict__ x, const float* __restrict__ w,
    const float* __restrict__ mask, bf16* __restrict__ out) {
  const int m = blockIdx.x;
  const float* row = x + (size_t)m * DMODEL;
  float v[8];
  float ss = 0.f;
#pragma unroll
  for (int i = 0; i < 8; ++i) {
    v[i] = row[threadIdx.x + i * 256];
    ss += v[i] * v[i];
  }
#pragma unroll
  for (int o = 16; o > 0; o >>= 1) ss += __shfl_xor(ss, o, 32);
  __shared__ float red[8];
  if ((threadIdx.x & 31) == 0) red[threadIdx.x >> 5] = ss;
  __syncthreads();
  float tot = red[0] + red[1] + red[2] + red[3] + red[4] + red[5] + red[6] + red[7];
  float rinv = rsqrtf(tot * (1.0f / (float)DMODEL) + 1e-6f);
  float msc = mask ? mask[m] : 1.f;
#pragma unroll
  for (int i = 0; i < 8; ++i) {
    int d = threadIdx.x + i * 256;
    out[(size_t)m * DMODEL + d] = f2bf(v[i] * rinv * w[d] * msc);
  }
}

// ---------------------------------------------------------------------------
// Tiled GEMM: C[M][N] (f32) = A[M][K] (bf16) * W[Npad][K]^T (bf16)
// EPI==1 additionally adds Res[M][N] before the store.
// 128x128 block tile, 8 waves (4 Mx2 N groups), 2x4 WMMA tiles per wave.
// TDM double-buffered staging: wave 0 issues tensor_load_to_lds for the next
// K-tile, waits TENSORcnt, and the block barrier publishes the data.
// ---------------------------------------------------------------------------
template <int EPI>
__global__ __launch_bounds__(256) void gemm_kernel(
    const bf16* __restrict__ A, const bf16* __restrict__ Bw,
    const float* __restrict__ Res, float* __restrict__ Cout,
    int M, int N, int K) {
  __shared__ bf16 As[2][128 * LDSS];
  __shared__ bf16 Bs[2][128 * LDSS];

  const int tid  = threadIdx.x;
  const int lane = tid & 31;
  const int wave = tid >> 5;
  const int wm   = wave & 3;   // 4 groups of 32 rows
  const int wn   = wave >> 2;  // 2 groups of 64 cols
  const int lr   = lane & 15;
  const int hbase = (lane < 16) ? 0 : 8;
  const int ak0   = (lane < 16) ? 0 : 16;
  const int bk0   = (lane < 16) ? 0 : 16;

  const int m0 = blockIdx.y * 128;
  const int n0 = blockIdx.x * 128;
  const bool dma = (wave == 0);

  v8f acc[2][4];
#pragma unroll
  for (int mt = 0; mt < 2; ++mt)
#pragma unroll
    for (int nt = 0; nt < 4; ++nt) acc[mt][nt] = (v8f)(0.f);

  const int nk = K / 32;
  if (dma) {
    tdm_load_tile(lds_addr_of(&As[0][0]), A  + (size_t)m0 * K, (unsigned)K, (unsigned)M);
    tdm_load_tile(lds_addr_of(&Bs[0][0]), Bw + (size_t)n0 * K, (unsigned)K, (unsigned)M);
  }

  for (int i = 0; i < nk; ++i) {
    const int cur = i & 1;
    __syncthreads();                     // buf[cur^1] readers (iter i-1) done
    if (dma) {
      if (i + 1 < nk) {
        const int k1 = (i + 1) * 32;
        tdm_load_tile(lds_addr_of(&As[cur ^ 1][0]), A  + (size_t)m0 * K + k1,
                      (unsigned)K, (unsigned)M);
        tdm_load_tile(lds_addr_of(&Bs[cur ^ 1][0]), Bw + (size_t)n0 * K + k1,
                      (unsigned)K, (unsigned)M);
        __builtin_amdgcn_s_wait_tensorcnt(2);   // tile i complete (in-order)
      } else {
        __builtin_amdgcn_s_wait_tensorcnt(0);
      }
    }
    __syncthreads();                     // tile i visible to all waves

    v16bf afrag[2];
#pragma unroll
    for (int mt = 0; mt < 2; ++mt) {
      const bf16* ap = &As[cur][(wm * 32 + mt * 16 + lr) * LDSS + ak0];
      afrag[mt] = combine16(*(const v8bf*)ap, *(const v8bf*)(ap + 8));
    }
#pragma unroll
    for (int nt = 0; nt < 4; ++nt) {
      const bf16* bp = &Bs[cur][(wn * 64 + nt * 16 + lr) * LDSS + bk0];
      v16bf bfrag = combine16(*(const v8bf*)bp, *(const v8bf*)(bp + 8));
#pragma unroll
      for (int mt = 0; mt < 2; ++mt)
        acc[mt][nt] = __builtin_amdgcn_wmma_f32_16x16x32_bf16(
            false, afrag[mt], false, bfrag, (short)0, acc[mt][nt], false, false);
    }
  }

#pragma unroll
  for (int mt = 0; mt < 2; ++mt)
#pragma unroll
    for (int nt = 0; nt < 4; ++nt) {
      int col = n0 + wn * 64 + nt * 16 + lr;
      if (col < N) {
#pragma unroll
        for (int r = 0; r < 8; ++r) {
          int rowi = m0 + wm * 32 + mt * 16 + hbase + r;
          size_t idx = (size_t)rowi * N + col;
          float v = acc[mt][nt][r];
          if (EPI == 1) v += Res[idx];
          Cout[idx] = v;
        }
      }
    }
}

// ---------------------------------------------------------------------------
// Fused gate/up GEMM: act[M][N] (bf16) = silu(A*Wg^T) * (A*Wu^T)
// Same tiling; three TDM-staged tiles per K-step, double-buffered.
// ---------------------------------------------------------------------------
__global__ __launch_bounds__(256) void gateup_kernel(
    const bf16* __restrict__ A, const bf16* __restrict__ Wg,
    const bf16* __restrict__ Wu, bf16* __restrict__ act,
    int M, int N, int K) {
  __shared__ bf16 As[2][128 * LDSS];
  __shared__ bf16 Gs[2][128 * LDSS];
  __shared__ bf16 Us[2][128 * LDSS];

  const int tid  = threadIdx.x;
  const int lane = tid & 31;
  const int wave = tid >> 5;
  const int wm   = wave & 3;
  const int wn   = wave >> 2;
  const int lr   = lane & 15;
  const int hbase = (lane < 16) ? 0 : 8;
  const int ak0   = (lane < 16) ? 0 : 16;
  const int bk0   = (lane < 16) ? 0 : 16;

  const int m0 = blockIdx.y * 128;
  const int n0 = blockIdx.x * 128;
  const bool dma = (wave == 0);

  v8f accg[2][4], accu[2][4];
#pragma unroll
  for (int mt = 0; mt < 2; ++mt)
#pragma unroll
    for (int nt = 0; nt < 4; ++nt) { accg[mt][nt] = (v8f)(0.f); accu[mt][nt] = (v8f)(0.f); }

  const int nk = K / 32;
  if (dma) {
    tdm_load_tile(lds_addr_of(&As[0][0]), A  + (size_t)m0 * K, (unsigned)K, (unsigned)M);
    tdm_load_tile(lds_addr_of(&Gs[0][0]), Wg + (size_t)n0 * K, (unsigned)K, (unsigned)N);
    tdm_load_tile(lds_addr_of(&Us[0][0]), Wu + (size_t)n0 * K, (unsigned)K, (unsigned)N);
  }

  for (int i = 0; i < nk; ++i) {
    const int cur = i & 1;
    __syncthreads();
    if (dma) {
      if (i + 1 < nk) {
        const int k1 = (i + 1) * 32;
        tdm_load_tile(lds_addr_of(&As[cur ^ 1][0]), A  + (size_t)m0 * K + k1,
                      (unsigned)K, (unsigned)M);
        tdm_load_tile(lds_addr_of(&Gs[cur ^ 1][0]), Wg + (size_t)n0 * K + k1,
                      (unsigned)K, (unsigned)N);
        tdm_load_tile(lds_addr_of(&Us[cur ^ 1][0]), Wu + (size_t)n0 * K + k1,
                      (unsigned)K, (unsigned)N);
        __builtin_amdgcn_s_wait_tensorcnt(3);
      } else {
        __builtin_amdgcn_s_wait_tensorcnt(0);
      }
    }
    __syncthreads();

    v16bf afrag[2];
#pragma unroll
    for (int mt = 0; mt < 2; ++mt) {
      const bf16* aq = &As[cur][(wm * 32 + mt * 16 + lr) * LDSS + ak0];
      afrag[mt] = combine16(*(const v8bf*)aq, *(const v8bf*)(aq + 8));
    }
#pragma unroll
    for (int nt = 0; nt < 4; ++nt) {
      const bf16* gq = &Gs[cur][(wn * 64 + nt * 16 + lr) * LDSS + bk0];
      v16bf gfrag = combine16(*(const v8bf*)gq, *(const v8bf*)(gq + 8));
#pragma unroll
      for (int mt = 0; mt < 2; ++mt)
        accg[mt][nt] = __builtin_amdgcn_wmma_f32_16x16x32_bf16(
            false, afrag[mt], false, gfrag, (short)0, accg[mt][nt], false, false);
      const bf16* uq = &Us[cur][(wn * 64 + nt * 16 + lr) * LDSS + bk0];
      v16bf ufrag = combine16(*(const v8bf*)uq, *(const v8bf*)(uq + 8));
#pragma unroll
      for (int mt = 0; mt < 2; ++mt)
        accu[mt][nt] = __builtin_amdgcn_wmma_f32_16x16x32_bf16(
            false, afrag[mt], false, ufrag, (short)0, accu[mt][nt], false, false);
    }
  }

#pragma unroll
  for (int mt = 0; mt < 2; ++mt)
#pragma unroll
    for (int nt = 0; nt < 4; ++nt) {
      int col = n0 + wn * 64 + nt * 16 + lr;
      if (col < N) {
#pragma unroll
        for (int r = 0; r < 8; ++r) {
          int rowi = m0 + wm * 32 + mt * 16 + hbase + r;
          float g = accg[mt][nt][r];
          float u = accu[mt][nt][r];
          float s = g / (1.f + __expf(-g));
          act[(size_t)rowi * N + col] = f2bf(s * u);
        }
      }
    }
}

// ---------------------------------------------------------------------------
// Depthwise causal conv over time (width 4) on the xBC slice of zxbcdt.
// ---------------------------------------------------------------------------
__global__ void conv_kernel(const float* __restrict__ zxbcdt,
                            const float* __restrict__ conv_w,
                            const float* __restrict__ conv_b,
                            float* __restrict__ xBC) {
  size_t idx = (size_t)blockIdx.x * 256 + threadIdx.x;
  if (idx >= (size_t)MTOK * CONVD) return;
  int c = (int)(idx % CONVD);
  int m = (int)(idx / CONVD);
  int b = m / SEQ, l = m % SEQ;
  float acc = conv_b[c];
#pragma unroll
  for (int j = 0; j < 4; ++j) {
    int ll = l - 3 + j;
    if (ll >= 0)
      acc += conv_w[c * 4 + j] * zxbcdt[(size_t)(b * SEQ + ll) * DINP + DMODEL + c];
  }
  xBC[(size_t)m * CONVD + c] = acc;
}

// ---------------------------------------------------------------------------
// SSM scan. One block per (batch, head); thread (p=tid>>2, nq=tid&3) owns 16
// state entries in registers. Fuses +D*x and SiLU(z+z_bias) gating.
// ---------------------------------------------------------------------------
__global__ __launch_bounds__(256) void scan_kernel(
    const float* __restrict__ zxbcdt, const float* __restrict__ xBC,
    const float* __restrict__ Dvec, const float* __restrict__ z_bias,
    bf16* __restrict__ gated) {
  const int bh = blockIdx.x;
  const int b = bh >> 5, head = bh & 31;
  const int tid = threadIdx.x;
  const int p = tid >> 2;
  const int nq = tid & 3;
  const int n0 = nq * 16;

  float hst[16];
#pragma unroll
  for (int i = 0; i < 16; ++i) hst[i] = 0.f;
  const float Dh = Dvec[head];

  __shared__ float sx[64], sB[64], sC[64];

  for (int l = 0; l < SEQ; ++l) {
    const size_t m = (size_t)b * SEQ + l;
    const float* xr = xBC + m * CONVD;
    if (tid < 64)        sx[tid]       = xr[head * 64 + tid];
    else if (tid < 128)  sB[tid - 64]  = xr[DMODEL + head * 64 + (tid - 64)];
    else if (tid < 192)  sC[tid - 128] = xr[DMODEL + NHEADS * DSTATE + head * 64 + (tid - 128)];
    float alog = zxbcdt[m * DINP + DMODEL + CONVD + head];
    float dt = (alog > 20.f) ? alog : log1pf(__expf(alog));   // softplus
    float a = __expf(-dt);
    __syncthreads();

    float xp = sx[p];
    float y = 0.f;
#pragma unroll
    for (int i = 0; i < 16; ++i) {
      float hn = a * hst[i] + xp * sB[n0 + i];
      hst[i] = hn;
      y += hn * sC[n0 + i];
    }
    y += __shfl_xor(y, 1, 32);
    y += __shfl_xor(y, 2, 32);
    if (nq == 0) {
      y += Dh * xp;
      int d = head * 64 + p;
      float z = zxbcdt[m * DINP + d] + z_bias[d];
      float sz = z / (1.f + __expf(-z));
      gated[m * DMODEL + d] = f2bf(y * sz);
    }
    __syncthreads();
  }
}

// ---------------------------------------------------------------------------
// Host-side orchestration
// ---------------------------------------------------------------------------
extern "C" void kernel_launch(void* const* d_in, const int* in_sizes, int n_in,
                              void* d_out, int out_size, void* d_ws, size_t ws_size,
                              hipStream_t stream) {
  (void)in_sizes; (void)n_in; (void)out_size; (void)ws_size;
  const float* hidden = (const float*)d_in[0];
  const float* mask   = (const float*)d_in[1];
  const float* w_in   = (const float*)d_in[2];
  const float* conv_w = (const float*)d_in[3];
  const float* conv_b = (const float*)d_in[4];
  const float* z_bias = (const float*)d_in[5];
  const float* Dvec   = (const float*)d_in[6];
  const float* w_out  = (const float*)d_in[7];
  const float* ln1    = (const float*)d_in[8];
  const float* ln2    = (const float*)d_in[9];
  const float* w_gate = (const float*)d_in[10];
  const float* w_up   = (const float*)d_in[11];
  const float* w_down = (const float*)d_in[12];
  float* out = (float*)d_out;

  char* ws = (char*)d_ws;
  auto carve = [&](size_t bytes) {
    char* p = ws;
    ws += (bytes + 255) & ~(size_t)255;
    return p;
  };
  bf16*  w_in_b   = (bf16*)carve((size_t)DINP_PAD * DMODEL * 2);
  bf16*  w_out_b  = (bf16*)carve((size_t)DMODEL * DMODEL * 2);
  bf16*  w_gate_b = (bf16*)carve((size_t)INTER_N * DMODEL * 2);
  bf16*  w_up_b   = (bf16*)carve((size_t)INTER_N * DMODEL * 2);
  bf16*  w_down_b = (bf16*)carve((size_t)DMODEL * INTER_N * 2);
  bf16*  h1_b     = (bf16*)carve((size_t)MTOK * DMODEL * 2);
  float* zxbcdt   = (float*)carve((size_t)MTOK * DINP * 4);
  float* xBC      = (float*)carve((size_t)MTOK * CONVD * 4);
  bf16*  gated_b  = (bf16*)carve((size_t)MTOK * DMODEL * 2);
  float* hmid     = (float*)carve((size_t)MTOK * DMODEL * 4);
  bf16*  h2_b     = (bf16*)carve((size_t)MTOK * DMODEL * 2);
  bf16*  act_b    = (bf16*)carve((size_t)MTOK * INTER_N * 2);

  auto cvtgrid = [](size_t n) { return (unsigned)((n + 255) / 256); };

  // 1) weights f32 -> bf16 (padded rows)
  cvt_pad_kernel<<<cvtgrid((size_t)DINP_PAD * DMODEL), 256, 0, stream>>>(w_in,   w_in_b,   DINP,   DMODEL,  DINP_PAD);
  cvt_pad_kernel<<<cvtgrid((size_t)DMODEL   * DMODEL), 256, 0, stream>>>(w_out,  w_out_b,  DMODEL, DMODEL,  DMODEL);
  cvt_pad_kernel<<<cvtgrid((size_t)INTER_N  * DMODEL), 256, 0, stream>>>(w_gate, w_gate_b, INTER_N, DMODEL, INTER_N);
  cvt_pad_kernel<<<cvtgrid((size_t)INTER_N  * DMODEL), 256, 0, stream>>>(w_up,   w_up_b,   INTER_N, DMODEL, INTER_N);
  cvt_pad_kernel<<<cvtgrid((size_t)DMODEL   * INTER_N), 256, 0, stream>>>(w_down, w_down_b, DMODEL, INTER_N, DMODEL);

  // 2) RMSNorm + mask -> bf16
  rmsnorm_kernel<<<MTOK, 256, 0, stream>>>(hidden, ln1, mask, h1_b);

  // 3) in_proj GEMM: zxbcdt[4096][8224]
  gemm_kernel<0><<<dim3(DINP_PAD / 128, MTOK / 128), 256, 0, stream>>>(
      h1_b, w_in_b, nullptr, zxbcdt, MTOK, DINP, DMODEL);

  // 4) depthwise causal conv on xBC slice
  conv_kernel<<<cvtgrid((size_t)MTOK * CONVD), 256, 0, stream>>>(zxbcdt, conv_w, conv_b, xBC);

  // 5) SSM scan + D*x + SiLU(z) gating -> bf16
  scan_kernel<<<BATCH * NHEADS, 256, 0, stream>>>(zxbcdt, xBC, Dvec, z_bias, gated_b);

  // 6) out_proj GEMM + residual: hmid = gated @ Wout^T + hidden
  gemm_kernel<1><<<dim3(DMODEL / 128, MTOK / 128), 256, 0, stream>>>(
      gated_b, w_out_b, hidden, hmid, MTOK, DMODEL, DMODEL);

  // 7) second RMSNorm -> bf16
  rmsnorm_kernel<<<MTOK, 256, 0, stream>>>(hmid, ln2, nullptr, h2_b);

  // 8) fused gate/up GEMM + SiLU-mul -> bf16 act
  gateup_kernel<<<dim3(INTER_N / 128, MTOK / 128), 256, 0, stream>>>(
      h2_b, w_gate_b, w_up_b, act_b, MTOK, INTER_N, DMODEL);

  // 9) down GEMM + residual -> output (f32)
  gemm_kernel<1><<<dim3(DMODEL / 128, MTOK / 128), 256, 0, stream>>>(
      act_b, w_down_b, hmid, out, MTOK, DMODEL, INTER_N);
}
